// CondAttLSTM_38517266711125
// MI455X (gfx1250) — matrix operand order repaired and checked by reading, not compiled, not run on hardware
//
#include <hip/hip_runtime.h>
#include <hip/hip_bf16.h>
#include <math.h>

// ---------------- problem dimensions ----------------
#define B_   32
#define T_   128
#define I_   1024
#define H_   1024
#define C_   1024
#define L_   128
#define A_   512
#define NG   4096   // 4*H : packed gate width (input|forget|cell|out)
#define KCAT 3072   // 3*H : [h | ctx_vec | par_h]

typedef __attribute__((ext_vector_type(16))) __bf16 v16bf;
typedef __attribute__((ext_vector_type(8)))  float  v8f;
typedef __attribute__((ext_vector_type(4)))  int    v4i;

#ifndef __has_builtin
#define __has_builtin(x) 0
#endif
#if __has_builtin(__builtin_amdgcn_global_load_async_to_lds_b128) && \
    __has_builtin(__builtin_amdgcn_s_wait_asynccnt)
#define USE_ASYNC_LDS 1
#else
#define USE_ASYNC_LDS 0
#endif

#if USE_ASYNC_LDS
// per-lane async copy global -> LDS (tracked on ASYNCcnt)
#define GLD_ASYNC_B128(gp, lp)                                              \
    __builtin_amdgcn_global_load_async_to_lds_b128(                         \
        (__attribute__((address_space(1))) v4i*)(gp),                       \
        (__attribute__((address_space(3))) v4i*)(lp), 0, 0)
#endif

// float -> bf16 (round to nearest even), stored as raw u16
__device__ __forceinline__ unsigned short f2bf(float f) {
    union { float f; unsigned u; } v; v.f = f;
    unsigned r = v.u + 0x7FFFu + ((v.u >> 16) & 1u);
    return (unsigned short)(r >> 16);
}

// B-side tiled index: (32k x 16n) tiles, [n][k]-major inside a tile.
// element (k,n) of a [K,N] matrix -> ((k/32)*(N/16) + n/16)*512 + (n%16)*32 + (k%32)
__device__ __forceinline__ size_t btile_idx(int k, int n, int N) {
    return ((size_t)(k >> 5) * (N >> 4) + (n >> 4)) * 512 + (size_t)(n & 15) * 32 + (k & 31);
}

// ---------------- pack / convert kernels ----------------

// X [B,T,I] f32 -> Xb [T*B, I] bf16 row-major (A-side, time-major rows)
__global__ void k_pack_X(const float* __restrict__ X, unsigned short* __restrict__ Xb) {
    int idx = blockIdx.x * 256 + threadIdx.x;          // over T*B*I
    int k  = idx % I_;
    int tb = idx / I_;
    int b  = tb % B_;
    int t  = tb / B_;
    Xb[idx] = f2bf(X[((size_t)b * T_ + t) * I_ + k]);
}

// generic f32 -> bf16 copy (A-side, row-major preserved)
__global__ void k_cvt(const float* __restrict__ src, unsigned short* __restrict__ dst, int n) {
    int i = blockIdx.x * 256 + threadIdx.x;
    if (i < n) dst[i] = f2bf(src[i]);
}

// f32 [K,N] row-major -> bf16 B-tiled layout
__global__ void k_cvt_bt(const float* __restrict__ src, unsigned short* __restrict__ dst,
                         int K, int N) {
    int i = blockIdx.x * 256 + threadIdx.x;            // over K*N
    if (i >= K * N) return;
    int n = i % N, k = i / N;
    dst[btile_idx(k, n, N)] = f2bf(src[i]);
}

// pack 4 gate matrices [K,H] -> [K, 4H] bf16 in B-tiled layout
__global__ void k_pack_w4(const float* __restrict__ w0, const float* __restrict__ w1,
                          const float* __restrict__ w2, const float* __restrict__ w3,
                          unsigned short* __restrict__ dst, int K) {
    int i = blockIdx.x * 256 + threadIdx.x;            // over K*NG
    if (i >= K * NG) return;
    int n = i % NG, k = i / NG;
    int g = n >> 10, j = n & 1023;
    const float* w = (g == 0) ? w0 : (g == 1) ? w1 : (g == 2) ? w2 : w3;
    dst[btile_idx(k, n, NG)] = f2bf(w[(size_t)k * H_ + j]);
}

// pack 4 bias vectors [H] -> [4H] f32
__global__ void k_pack_b4(const float* __restrict__ b0, const float* __restrict__ b1,
                          const float* __restrict__ b2, const float* __restrict__ b3,
                          float* __restrict__ dst) {
    int i = blockIdx.x * 256 + threadIdx.x;            // NG
    int g = i >> 10, j = i & 1023;
    const float* bp = (g == 0) ? b0 : (g == 1) ? b1 : (g == 2) ? b2 : b3;
    dst[i] = bp[j];
}

// ---------------- WMMA bf16 GEMM (C = A*B [+addend] [+biasN]) ----------------
// A: [M,K] bf16 row-major, leading dim lda (contiguous fragment runs)
// B: bf16 in B-tiled layout (btile_idx) for [K,N]
// C/addend: [M,N] f32 row-major
// Block = 256 threads = 8 waves. One 16-row A slab shared per block; each wave
// owns a 16x64 output strip (4 WMMA tiles): one A fragment feeds 4 WMMAs.
// Requires M%16==0, N%512==0, K%32==0.
__global__ void k_wmma_gemm(const unsigned short* __restrict__ A, int lda,
                            const unsigned short* __restrict__ Bm,
                            float* __restrict__ Cm,
                            int M, int N, int K,
                            const float* __restrict__ addend,
                            const float* __restrict__ biasN) {
    __shared__ __align__(16) unsigned short smA[16 * 32];
    __shared__ __align__(16) unsigned short smB[8][4 * 512];   // 4 n-tiles per wave

    const int tid  = threadIdx.x;
    const int wave = tid >> 5;
    const int lane = tid & 31;
    const int rowBase = blockIdx.y * 16;
    const int n0 = (blockIdx.x * 8 + wave) * 64;
    const int ntiles = N >> 4;

    v8f acc[4];
    #pragma unroll
    for (int s = 0; s < 4; ++s) acc[s] = (v8f){0.f,0.f,0.f,0.f,0.f,0.f,0.f,0.f};

    for (int k0 = 0; k0 < K; k0 += 32) {
        // this wave's 4 n-tiles for k-block k0 are 4 KiB CONTIGUOUS in memory
        const unsigned short* bsrc =
            Bm + ((size_t)(k0 >> 5) * ntiles + (n0 >> 4)) * 512;
        __syncthreads();
#if USE_ASYNC_LDS
        if (wave == 0) {                    // A slab 16x32 (1 KiB) staged by wave 0
            #pragma unroll
            for (int i = 0; i < 2; ++i) {
                int idx = i * 32 + lane, r = idx >> 2, c = idx & 3;
                GLD_ASYNC_B128(&A[(size_t)(rowBase + r) * lda + k0 + c * 8],
                               &smA[r * 32 + c * 8]);
            }
        }
        #pragma unroll
        for (int i = 0; i < 8; ++i) {       // 4 KiB contiguous strip per wave
            int off = (i * 32 + lane) * 8;
            GLD_ASYNC_B128(bsrc + off, &smB[wave][off]);
        }
        __builtin_amdgcn_s_wait_asynccnt(0);
#else
        if (wave == 0) {
            #pragma unroll
            for (int i = 0; i < 2; ++i) {
                int idx = i * 32 + lane, r = idx >> 2, c = idx & 3;
                *(uint4*)(&smA[r * 32 + c * 8]) =
                    *(const uint4*)(&A[(size_t)(rowBase + r) * lda + k0 + c * 8]);
            }
        }
        #pragma unroll
        for (int i = 0; i < 8; ++i) {
            int off = (i * 32 + lane) * 8;
            *(uint4*)(&smB[wave][off]) = *(const uint4*)(bsrc + off);
        }
#endif
        if (k0 + 32 < K) {   // hint next K-slab toward the caches (global_prefetch_b8)
            __builtin_prefetch(&A[(size_t)(rowBase + (lane & 15)) * lda + k0 + 32], 0, 0);
            __builtin_prefetch(Bm + ((size_t)((k0 + 32) >> 5) * ntiles + (n0 >> 4)) * 512
                                  + lane * 8, 0, 0);
        }
        __syncthreads();

        // fragments in contiguous k order (k = 16*half + s) for both A and B — all loads are aligned b128 runs
        union Frag { v16bf v; unsigned short u[16]; uint4 q[2]; };
        Frag fa, fb;
        const int mrow  = lane & 15;
        const int ncol  = lane & 15;
        const int aHalf = (lane >> 4) << 4;   // A K-half select (0 / 16)
        const int bHalf = (lane >> 4) << 4;   // B K-half select (0 / 16)
        // A: k run [aHalf, aHalf+16)
        fa.q[0] = *(const uint4*)(&smA[mrow * 32 + aHalf]);
        fa.q[1] = *(const uint4*)(&smA[mrow * 32 + aHalf + 8]);
        #pragma unroll
        for (int s = 0; s < 4; ++s) {         // one A fragment feeds 4 WMMAs
            const unsigned short* bp = &smB[wave][s * 512 + ncol * 32 + bHalf];
            fb.q[0] = *(const uint4*)(bp);
            fb.q[1] = *(const uint4*)(bp + 8);
            acc[s] = __builtin_amdgcn_wmma_f32_16x16x32_bf16(
                         false, fa.v, false, fb.v, (short)0, acc[s], false, false);
        }
    }

    // epilogue: C/D layout — VGPR e holds M = e + 8*(lane>=16), N = lane&15
    #pragma unroll
    for (int s = 0; s < 4; ++s) {
        #pragma unroll
        for (int e = 0; e < 8; ++e) {
            int m = ((lane >> 4) << 3) + e;
            int n = s * 16 + (lane & 15);
            size_t row = rowBase + m, col = n0 + n;
            float v = acc[s][e];
            if (addend) v += addend[row * N + col];
            if (biasN)  v += biasN[col];
            Cm[row * N + col] = v;
        }
    }
}

// ---------------- attention: scores, softmax + context contraction ----------------

// scores[b,l] = sum_a tanh(ctx_pre[b,l,a] + hW[b,a]) * w_att[a] + b_att
__global__ void k_attn_scores(const float* __restrict__ ctx_pre, const float* __restrict__ hW,
                              const float* __restrict__ w_att, const float* __restrict__ b_att,
                              float* __restrict__ scores) {
    int bl  = blockIdx.x;            // b*L + l
    int b   = bl / L_;
    int tid = threadIdx.x;           // 256
    float p = 0.f;
    for (int a = tid; a < A_; a += 256)
        p += tanhf(ctx_pre[(size_t)bl * A_ + a] + hW[(size_t)b * A_ + a]) * w_att[a];
    for (int off = 16; off > 0; off >>= 1) p += __shfl_down(p, off, 32);
    __shared__ float red[8];
    if ((tid & 31) == 0) red[tid >> 5] = p;
    __syncthreads();
    if (tid == 0) {
        float s = 0.f;
        for (int w = 0; w < 8; ++w) s += red[w];
        scores[bl] = s + b_att[0];
    }
}

// softmax over L, then ctx_vec[b,:] = sum_l alpha_l * context[b,l,:]
__global__ void k_softmax_ctx(const float* __restrict__ scores, const float* __restrict__ context,
                              float* __restrict__ ctxs_out, unsigned short* __restrict__ hcat, int t) {
    int b = blockIdx.x;
    int tid = threadIdx.x;           // 256
    __shared__ float al[L_];
    if (tid < L_) al[tid] = scores[b * L_ + tid];
    __syncthreads();
    float mx = -1e30f;
    for (int l = 0; l < L_; ++l) mx = fmaxf(mx, al[l]);
    float sum = 0.f;
    for (int l = 0; l < L_; ++l) sum += expf(al[l] - mx);
    __syncthreads();
    if (tid < L_) al[tid] = expf(al[tid] - mx) / sum;
    __syncthreads();
    for (int c = tid; c < C_; c += 256) {
        float acc = 0.f;
        for (int l = 0; l < L_; ++l)
            acc += al[l] * context[((size_t)b * L_ + l) * C_ + c];
        ctxs_out[((size_t)b * T_ + t) * C_ + c] = acc;
        hcat[b * KCAT + H_ + c] = f2bf(acc);   // middle third of fused A operand
    }
}

// parent hidden gather into hcat (zero at t==0, matching reference)
__global__ void k_parent(const int* __restrict__ parent_t, const unsigned short* __restrict__ hist,
                         unsigned short* __restrict__ hcat, int t) {
    int b = blockIdx.x;
    int tid = threadIdx.x;           // 256
    int par = parent_t[b * T_ + t];
    for (int j = tid; j < H_; j += 256) {
        unsigned short v = (t == 0) ? (unsigned short)0
                                    : hist[((size_t)b * T_ + par) * H_ + j];
        hcat[b * KCAT + 2 * H_ + j] = v;
    }
}

// pointwise LSTM cell update; writes outputs + next-step bf16 state
__global__ void k_lstm_update(const float* __restrict__ g, float* __restrict__ c_state,
                              float* __restrict__ hs, float* __restrict__ cs,
                              unsigned short* __restrict__ hcat, unsigned short* __restrict__ hist,
                              int t) {
    int idx = blockIdx.x * 256 + threadIdx.x;   // B*H
    int b = idx >> 10, j = idx & 1023;
    float gi = g[(size_t)b * NG + j];
    float gf = g[(size_t)b * NG + H_ + j];
    float gc = g[(size_t)b * NG + 2 * H_ + j];
    float go = g[(size_t)b * NG + 3 * H_ + j];
    float i_t = 1.f / (1.f + expf(-gi));
    float f_t = 1.f / (1.f + expf(-gf));
    float o_t = 1.f / (1.f + expf(-go));
    float cn  = f_t * c_state[idx] + i_t * tanhf(gc);
    float hn  = o_t * tanhf(cn);
    c_state[idx] = cn;
    hs[((size_t)b * T_ + t) * H_ + j] = hn;
    cs[((size_t)b * T_ + t) * H_ + j] = cn;
    unsigned short hb = f2bf(hn);
    hcat[b * KCAT + j] = hb;
    hist[((size_t)b * T_ + t) * H_ + j] = hb;
}

// ---------------- host launcher ----------------
extern "C" void kernel_launch(void* const* d_in, const int* in_sizes, int n_in,
                              void* d_out, int out_size, void* d_ws, size_t ws_size,
                              hipStream_t stream) {
    (void)in_sizes; (void)n_in; (void)out_size; (void)ws_size;

    const float* X        = (const float*)d_in[0];
    const float* context  = (const float*)d_in[1];
    const int*   parent   = (const int*)  d_in[2];
    const float* w_in[4]  = {(const float*)d_in[3],  (const float*)d_in[4],
                             (const float*)d_in[5],  (const float*)d_in[6]};
    const float* u_in[4]  = {(const float*)d_in[7],  (const float*)d_in[8],
                             (const float*)d_in[9],  (const float*)d_in[10]};
    const float* c_in[4]  = {(const float*)d_in[11], (const float*)d_in[12],
                             (const float*)d_in[13], (const float*)d_in[14]};
    const float* p_in[4]  = {(const float*)d_in[15], (const float*)d_in[16],
                             (const float*)d_in[17], (const float*)d_in[18]};
    const float* w_ctx_att   = (const float*)d_in[19];   // [C,A]
    const float* w_ctx_att_h = (const float*)d_in[20];   // [H,A]
    const float* w_att       = (const float*)d_in[21];   // [A,1]
    const float* b_in[4]  = {(const float*)d_in[22], (const float*)d_in[23],
                             (const float*)d_in[24], (const float*)d_in[25]};
    const float* ctx_att_b = (const float*)d_in[26];     // [A]
    const float* b_att     = (const float*)d_in[27];     // [1]

    float* hs   = (float*)d_out;
    float* cs   = hs + (size_t)B_ * T_ * H_;
    float* ctxs = cs + (size_t)B_ * T_ * H_;

    // workspace carve-up (256B aligned)
    char* p = (char*)d_ws;
    auto alloc = [&](size_t bytes) -> void* {
        void* r = p; p += (bytes + 255) & ~(size_t)255; return r;
    };
    unsigned short* Xb    = (unsigned short*)alloc((size_t)T_ * B_ * I_ * 2);
    unsigned short* Ctxb  = (unsigned short*)alloc((size_t)B_ * L_ * C_ * 2);
    unsigned short* Wx    = (unsigned short*)alloc((size_t)I_ * NG * 2);
    unsigned short* Wg    = (unsigned short*)alloc((size_t)KCAT * NG * 2);
    unsigned short* Wah   = (unsigned short*)alloc((size_t)H_ * A_ * 2);
    unsigned short* Wca   = (unsigned short*)alloc((size_t)C_ * A_ * 2);
    float*          bias4 = (float*)alloc((size_t)NG * 4);
    float*          xg    = (float*)alloc((size_t)T_ * B_ * NG * 4);   // 64 MiB
    float*          ctxpre= (float*)alloc((size_t)B_ * L_ * A_ * 4);
    unsigned short* hist  = (unsigned short*)alloc((size_t)B_ * T_ * H_ * 2);
    unsigned short* hcat  = (unsigned short*)alloc((size_t)B_ * KCAT * 2);
    float*          cst   = (float*)alloc((size_t)B_ * H_ * 4);
    float*          hW    = (float*)alloc((size_t)B_ * A_ * 4);
    float*          scores= (float*)alloc((size_t)B_ * L_ * 4);
    float*          graw  = (float*)alloc((size_t)B_ * NG * 4);

    // ---- one-time conversions / packing ----
    k_pack_X<<<(T_ * B_ * I_) / 256, 256, 0, stream>>>(X, Xb);
    k_cvt<<<(B_ * L_ * C_) / 256, 256, 0, stream>>>(context, Ctxb, B_ * L_ * C_);
    k_cvt_bt<<<(C_ * A_) / 256, 256, 0, stream>>>(w_ctx_att, Wca, C_, A_);
    k_cvt_bt<<<(H_ * A_) / 256, 256, 0, stream>>>(w_ctx_att_h, Wah, H_, A_);
    k_pack_w4<<<(I_ * NG) / 256, 256, 0, stream>>>(w_in[0], w_in[1], w_in[2], w_in[3], Wx, I_);
    k_pack_w4<<<(H_ * NG) / 256, 256, 0, stream>>>(u_in[0], u_in[1], u_in[2], u_in[3],
                                                   Wg + 0 * (size_t)H_ * NG, H_);
    k_pack_w4<<<(C_ * NG) / 256, 256, 0, stream>>>(c_in[0], c_in[1], c_in[2], c_in[3],
                                                   Wg + 1 * (size_t)H_ * NG, C_);
    k_pack_w4<<<(H_ * NG) / 256, 256, 0, stream>>>(p_in[0], p_in[1], p_in[2], p_in[3],
                                                   Wg + 2 * (size_t)H_ * NG, H_);
    k_pack_b4<<<NG / 256, 256, 0, stream>>>(b_in[0], b_in[1], b_in[2], b_in[3], bias4);

    (void)hipMemsetAsync(hist, 0, (size_t)B_ * T_ * H_ * 2, stream);
    (void)hipMemsetAsync(hcat, 0, (size_t)B_ * KCAT * 2, stream);
    (void)hipMemsetAsync(cst,  0, (size_t)B_ * H_ * 4, stream);

    // ---- big parallel GEMMs ----
    // xg[T*B, 4H] = Xb @ Wx + bias4
    k_wmma_gemm<<<dim3(NG / 512, (T_ * B_) / 16), 256, 0, stream>>>(
        Xb, I_, Wx, xg, T_ * B_, NG, I_, nullptr, bias4);
    // ctx_pre[B*L, A] = Ctxb @ Wca + ctx_att_b
    k_wmma_gemm<<<dim3(A_ / 512, (B_ * L_) / 16), 256, 0, stream>>>(
        Ctxb, C_, Wca, ctxpre, B_ * L_, A_, C_, nullptr, ctx_att_b);

    // ---- sequential recurrence ----
    for (int t = 0; t < T_; ++t) {
        // hW[B,A] = h @ w_context_att_h   (h = hcat cols [0,H))
        k_wmma_gemm<<<dim3(A_ / 512, B_ / 16), 256, 0, stream>>>(
            hcat, KCAT, Wah, hW, B_, A_, H_, nullptr, nullptr);
        k_attn_scores<<<B_ * L_, 256, 0, stream>>>(ctxpre, hW, w_att, b_att, scores);
        k_softmax_ctx<<<B_, 256, 0, stream>>>(scores, context, ctxs, hcat, t);
        k_parent<<<B_, 256, 0, stream>>>(parent, hist, hcat, t);
        // graw[B,4H] = [h|ctx|par_h] @ [u;c;p] + xg_t
        k_wmma_gemm<<<dim3(NG / 512, B_ / 16), 256, 0, stream>>>(
            hcat, KCAT, Wg, graw, B_, NG, KCAT, xg + (size_t)t * B_ * NG, nullptr);
        k_lstm_update<<<(B_ * H_) / 256, 256, 0, stream>>>(
            graw, cst, hs, cs, hcat, hist, t);
    }
}
